// CARAFE_55791625175243
// MI455X (gfx1250) — matrix operand order repaired and checked by reading, not compiled, not run on hardware
//
#include <hip/hip_runtime.h>

typedef __attribute__((ext_vector_type(16))) _Float16 v16h;
typedef __attribute__((ext_vector_type(8)))  _Float16 v8h;
typedef __attribute__((ext_vector_type(8)))  float    v8f;

#define BATCH 8
#define C1    192
#define CC    64
#define HW    4096      // 64*64
#define NPAD  112       // 100 enc channels padded to 7*16
#define KDIM  576       // 64*9
#define EPS   1e-3f

union AFrag { v16h v; v8h h[2]; };

// ---------------------------------------------------------------------------
// Kernel 1: fold BN into f16 weights + f32 biases.
//   wC[n*192+k] = comp_w[n,k] * gscale(n)          (1x1 conv is already [n][k])
//   wE[n*576+k], k = (dy*3+dx)*64 + c, padded with zeros for n in [100,112)
// ---------------------------------------------------------------------------
__global__ void carafe_prep(const float* __restrict__ comp_w,
                            const float* __restrict__ comp_g,
                            const float* __restrict__ comp_b,
                            const float* __restrict__ comp_m,
                            const float* __restrict__ comp_v,
                            const float* __restrict__ enc_w,
                            const float* __restrict__ enc_g,
                            const float* __restrict__ enc_b,
                            const float* __restrict__ enc_m,
                            const float* __restrict__ enc_v,
                            _Float16* __restrict__ wC, float* __restrict__ bC,
                            _Float16* __restrict__ wE, float* __restrict__ bE) {
  int gid = blockIdx.x * blockDim.x + threadIdx.x;
  if (gid < CC) {
    float sc = comp_g[gid] * rsqrtf(comp_v[gid] + EPS);
    bC[gid] = comp_b[gid] - comp_m[gid] * sc;
  }
  if (gid < NPAD) {
    float v = 0.0f;
    if (gid < 100) {
      float sc = enc_g[gid] * rsqrtf(enc_v[gid] + EPS);
      v = enc_b[gid] - enc_m[gid] * sc;
    }
    bE[gid] = v;
  }
  if (gid < CC * C1) {
    int n = gid / C1;
    float sc = comp_g[n] * rsqrtf(comp_v[n] + EPS);
    wC[gid] = (_Float16)(comp_w[gid] * sc);
  }
  int e = gid - CC * C1;
  if (e >= 0 && e < NPAD * KDIM) {
    int n = e / KDIM, k = e % KDIM;
    int tap = k >> 6, c = k & 63;
    int dy = tap / 3, dx = tap % 3;
    float v = 0.0f;
    if (n < 100) {
      float sc = enc_g[n] * rsqrtf(enc_v[n] + EPS);
      v = enc_w[((n * 64 + c) * 3 + dy) * 3 + dx] * sc;
    }
    wE[e] = (_Float16)v;
  }
}

// ---------------------------------------------------------------------------
// Kernel 2: compressor — 1x1 conv as WMMA GEMM (M=4096/batch, K=192, N=64)
//   + BN bias + SiLU, writes t as NHWC f16.
// Block: 128 threads (4 waves); each wave owns a 16-position M tile.
// ---------------------------------------------------------------------------
__global__ __launch_bounds__(128) void carafe_compress(
    const float* __restrict__ x, const _Float16* __restrict__ wC,
    const float* __restrict__ bC, _Float16* __restrict__ t) {
  __shared__ __align__(32) _Float16 sW[CC * C1];     // 24 KB, layout [n][k]
  __shared__ __align__(16) _Float16 sA[4][32 * 16];  // per-wave [m][k] tile

  const int tid  = threadIdx.x;
  const int wave = tid >> 5, lane = tid & 31;
  const int g = lane >> 4, ml = lane & 15;
  const int b  = blockIdx.y;
  const int p0 = blockIdx.x * 64 + wave * 16;

  { // stage all folded weights once (24576 B = 1536 uint4)
    const uint4* src = (const uint4*)wC;
    uint4* dst = (uint4*)sW;
    for (int i = tid; i < 1536; i += 128) dst[i] = src[i];
  }
  __syncthreads();

  const float* xb = x + (size_t)b * C1 * HW;
  v8f acc[4] = {};

  for (int kc = 0; kc < 6; ++kc) {
    const int k0 = kc * 32;
    // stage A tile: sA[wave][m*32+k] = x[b, k0+k, p0+m], coalesced along m
    for (int it = 0; it < 16; ++it) {
      int kk = it * 2 + g;
      float v = xb[(size_t)(k0 + kk) * HW + p0 + ml];
      sA[wave][ml * 32 + kk] = (_Float16)v;
    }
    __syncthreads();
    // A fragment: lane holds M=ml; elements 0-15 -> K=16g..16g+15 (contiguous)
    AFrag A;
    A.h[0] = *(const v8h*)&sA[wave][ml * 32 + 16 * g];
    A.h[1] = *(const v8h*)&sA[wave][ml * 32 + 16 * g + 8];
#pragma unroll
    for (int nt = 0; nt < 4; ++nt) {
      int n = nt * 16 + ml;  // B: N=lane%16, K=16*(lane/16)+h -> contiguous
      const v16h Bf = *(const v16h*)&sW[n * C1 + k0 + 16 * g];
      acc[nt] = __builtin_amdgcn_wmma_f32_16x16x32_f16(
          false, A.v, false, Bf, (short)0, acc[nt], false, false);
    }
    __syncthreads();
  }

  // epilogue: bias + SiLU, store NHWC f16
  _Float16* tb = t + ((size_t)b * HW + p0) * CC;
#pragma unroll
  for (int nt = 0; nt < 4; ++nt) {
    int c = nt * 16 + ml;
    float bias = bC[c];
#pragma unroll
    for (int r = 0; r < 8; ++r) {
      int m = r + 8 * g;                     // C/D: M = r + 8*(lane/16)
      float v = acc[nt][r] + bias;
      v = v / (1.0f + __expf(-v));           // SiLU
      tb[(size_t)m * CC + c] = (_Float16)v;
    }
  }
}

// ---------------------------------------------------------------------------
// Kernel 3: encoder 3x3 conv (im2col WMMA GEMM, K=576, N=112 padded)
//   + BN + pixelshuffle(2) + softmax(25) -> mask (b,25,128,128) f32.
// Block: 128 threads, 8x8 spatial tile; wave w owns rows [2w,2w+2).
// ---------------------------------------------------------------------------
__global__ __launch_bounds__(128) void carafe_encode(
    const _Float16* __restrict__ t, const _Float16* __restrict__ wE,
    const float* __restrict__ bE, float* __restrict__ mask) {
  __shared__ __align__(16) _Float16 sP[10 * 10 * CC];  // halo patch, 12.5 KB
  __shared__ float sM[4][16 * NPAD];                   // conv results, 28 KB

  const int tid  = threadIdx.x;
  const int wave = tid >> 5, lane = tid & 31;
  const int g = lane >> 4, ml = lane & 15;
  const int b  = blockIdx.y;
  const int by = (blockIdx.x >> 3) * 8, bx = (blockIdx.x & 7) * 8;

  // stage 10x10 halo patch (zero padded), 128 B per position
  for (int pos = tid; pos < 100; pos += 128) {
    int py = pos / 10, px = pos % 10;
    int y = by + py - 1, xx = bx + px - 1;
    uint4* dst = (uint4*)&sP[pos * CC];
    if ((unsigned)y < 64u && (unsigned)xx < 64u) {
      const uint4* src = (const uint4*)(t + ((size_t)b * HW + y * 64 + xx) * CC);
#pragma unroll
      for (int i = 0; i < 8; ++i) dst[i] = src[i];
    } else {
      uint4 z = make_uint4(0u, 0u, 0u, 0u);
#pragma unroll
      for (int i = 0; i < 8; ++i) dst[i] = z;
    }
  }
  __syncthreads();

  const int m  = ml;                        // A: M = lane%16
  const int my = wave * 2 + (m >> 3), mx = m & 7;
  v8f acc[7] = {};

  for (int kc = 0; kc < 18; ++kc) {
    const int k0 = kc * 32;                 // 32 channels of one 3x3 tap
    const int tap = k0 >> 6, c0 = k0 & 63;
    const int dy = tap / 3, dx = tap % 3;
    const int pbase = ((my + dy) * 10 + (mx + dx)) * CC + c0;
    AFrag A;
    A.h[0] = *(const v8h*)&sP[pbase + 16 * g];
    A.h[1] = *(const v8h*)&sP[pbase + 16 * g + 8];
#pragma unroll
    for (int nt = 0; nt < 7; ++nt) {
      int n = nt * 16 + ml;
      const v16h Bf = *(const v16h*)&wE[(size_t)n * KDIM + k0 + 16 * g];
      acc[nt] = __builtin_amdgcn_wmma_f32_16x16x32_f16(
          false, A.v, false, Bf, (short)0, acc[nt], false, false);
    }
  }

  // bias + park results in LDS for cross-lane softmax
#pragma unroll
  for (int nt = 0; nt < 7; ++nt) {
    int n = nt * 16 + ml;
    float bias = bE[n];
#pragma unroll
    for (int r = 0; r < 8; ++r)
      sM[wave][(r + 8 * g) * NPAD + n] = acc[nt][r] + bias;
  }
  __syncthreads();

  // softmax over 25 channels per (position, subpixel): 2 tasks per lane
  for (int tsk = 0; tsk < 2; ++tsk) {
    int task = lane * 2 + tsk;              // 0..63
    int pos = task >> 2, sub = task & 3;    // enc channel e = q*4 + sub
    const float* row = &sM[wave][pos * NPAD];
    float vals[25];
    float mx_ = -1e30f;
#pragma unroll
    for (int q = 0; q < 25; ++q) {
      vals[q] = row[q * 4 + sub];
      mx_ = fmaxf(mx_, vals[q]);
    }
    float s = 0.0f;
#pragma unroll
    for (int q = 0; q < 25; ++q) {
      vals[q] = __expf(vals[q] - mx_);
      s += vals[q];
    }
    float inv = 1.0f / s;
    int py = wave * 2 + (pos >> 3), px = pos & 7;
    int Y = 2 * (by + py) + (sub >> 1);
    int X = 2 * (bx + px) + (sub & 1);
    float* mb = mask + (size_t)b * 25 * 16384 + Y * 128 + X;
#pragma unroll
    for (int q = 0; q < 25; ++q) mb[(size_t)q * 16384] = vals[q] * inv;
  }
}

// ---------------------------------------------------------------------------
// Kernel 4: content-aware reassembly (memory bound).
// One thread per output pixel (b,Y,X); 25 mask weights + clipped offsets in
// registers, reused across all 192 channels.
// ---------------------------------------------------------------------------
__global__ __launch_bounds__(256) void carafe_reassemble(
    const float* __restrict__ x, const float* __restrict__ mask,
    float* __restrict__ out) {
  int gid = blockIdx.x * 256 + threadIdx.x;      // 8*128*128 threads
  int X = gid & 127, Y = (gid >> 7) & 127, b = gid >> 14;

  const float* mb = mask + (size_t)b * 25 * 16384 + Y * 128 + X;
  int y2 = Y >> 1, x2 = X >> 1;
  float w[25];
  int off[25];
#pragma unroll
  for (int i = 0; i < 5; ++i)
#pragma unroll
    for (int j = 0; j < 5; ++j) {
      int q = i * 5 + j;
      int sy = y2 + i - 2, sx = x2 + j - 2;      // floor((Y+2i-4)/2) etc.
      bool ok = ((unsigned)sy < 64u) && ((unsigned)sx < 64u);
      w[q]   = ok ? mb[(size_t)q * 16384] : 0.0f;  // zero-pad contributes 0
      off[q] = ok ? (sy * 64 + sx) : 0;
    }

  const float* xb = x + (size_t)b * C1 * HW;
  float* ob = out + (size_t)b * C1 * 16384 + Y * 128 + X;
  for (int c = 0; c < C1; ++c) {
    const float* xc = xb + (size_t)c * HW;
    __builtin_prefetch(xc + HW, 0, 0);           // global_prefetch next channel
    float acc = 0.0f;
#pragma unroll
    for (int q = 0; q < 25; ++q) acc += w[q] * xc[off[q]];
    ob[(size_t)c * 16384] = acc;
  }
}

// ---------------------------------------------------------------------------
extern "C" void kernel_launch(void* const* d_in, const int* in_sizes, int n_in,
                              void* d_out, int out_size, void* d_ws, size_t ws_size,
                              hipStream_t stream) {
  const float* x      = (const float*)d_in[0];
  const float* comp_w = (const float*)d_in[1];
  const float* comp_g = (const float*)d_in[2];
  const float* comp_b = (const float*)d_in[3];
  const float* comp_m = (const float*)d_in[4];
  const float* comp_v = (const float*)d_in[5];
  const float* enc_w  = (const float*)d_in[6];
  const float* enc_g  = (const float*)d_in[7];
  const float* enc_b  = (const float*)d_in[8];
  const float* enc_m  = (const float*)d_in[9];
  const float* enc_v  = (const float*)d_in[10];
  float* out = (float*)d_out;

  char* ws = (char*)d_ws;
  _Float16* wC  = (_Float16*)(ws + 0);                 // 24.0 KB
  float*    bC  = (float*)   (ws + (32u << 10));       // 256 B
  _Float16* wE  = (_Float16*)(ws + (64u << 10));       // 126 KB
  float*    bE  = (float*)   (ws + (256u << 10));      // 448 B
  _Float16* t   = (_Float16*)(ws + (320u << 10));      // 4 MB  (NHWC f16)
  float*    msk = (float*)   (ws + (5u << 20));        // 12.5 MB

  carafe_prep<<<300, 256, 0, stream>>>(comp_w, comp_g, comp_b, comp_m, comp_v,
                                       enc_w, enc_g, enc_b, enc_m, enc_v,
                                       wC, bC, wE, bE);

  dim3 gc(64, BATCH);
  carafe_compress<<<gc, 128, 0, stream>>>(x, wC, bC, t);

  dim3 ge(64, BATCH);
  carafe_encode<<<ge, 128, 0, stream>>>(t, wE, bE, msk);

  carafe_reassemble<<<512, 256, 0, stream>>>(x, msk, out);
}
